// DynamicReductionNetwork_17136919511679
// MI455X (gfx1250) — matrix operand order repaired and checked by reading, not compiled, not run on hardware
//
#include <hip/hip_runtime.h>
#include <math.h>

// ---------------- problem constants (match reference) ----------------
constexpr int   NODESN = 8192;     // B * NPG
constexpr int   NPGC   = 1024;
constexpr int   BG     = 8;
constexpr int   KNB    = 16;
constexpr int   HDIM   = 64;
constexpr int   EDIR   = NODESN * KNB;   // 131072 directed knn edges
constexpr int   ETOT   = 2 * EDIR;       // 262144 after symmetrization
constexpr float NEGF   = -1e30f;
constexpr float BIGF   = 1e30f;
constexpr int   CCH    = 512;            // kNN candidate chunk (LDS dist tile)

#define CDIV(a, b) (((a) + (b) - 1) / (b))

// ---------------- CDNA5 WMMA types & fragment helpers ----------------
typedef __attribute__((ext_vector_type(16))) __bf16 v16bf;
typedef __attribute__((ext_vector_type(8)))  __bf16 v8bf_t;
typedef __attribute__((ext_vector_type(4)))  __bf16 v4bf_t;
typedef __attribute__((ext_vector_type(8)))  float  v8f;
typedef __attribute__((ext_vector_type(4)))  float  v4f_t;

__device__ __forceinline__ v8f wmma_bf16(v16bf a, v16bf b, v8f c) {
  // D(16x16,f32) += A(16x32,bf16) * B(32x16,bf16)
  return __builtin_amdgcn_wmma_f32_16x16x32_bf16(false, a, false, b,
                                                 (short)0, c, false, false);
}

// A tile 16x32 (MxK) row-major bf16 in LDS, leading dim ld (multiple of 8).
// per lane the fragment is two contiguous 8-element runs
// (K kb..kb+7 and kb+8..kb+15, kb = kofs + 16*half) -> two ds_load_b128.
__device__ __forceinline__ v16bf load_a_frag(const __bf16* m, int ld, int kofs, int lane) {
  int row = lane & 15;
  int kb  = kofs + ((lane & 16) ? 16 : 0);
  const __bf16* p = m + row * ld + kb;
  v8bf_t lo = *(const v8bf_t*)(p);
  v8bf_t hi = *(const v8bf_t*)(p + 8);
  return __builtin_shufflevector(lo, hi, 0, 1, 2, 3, 4, 5, 6, 7,
                                 8, 9, 10, 11, 12, 13, 14, 15);
}

// B fragments are staged in fragment-major ("swizzled") LDS layout:
// frag base + lane*16 .. +15 contiguous -> single v16bf (2x ds_load_b128).
// Element (k, n) of a 32x16 KxN tile lives at lane = n + ((k&31)>=16 ? 16:0),
// v = k & 15 within its k/32 fragment of 512 elements.
__device__ __forceinline__ v16bf load_b_frag_sw(const __bf16* fragbase, int lane) {
  return *(const v16bf*)(fragbase + lane * 16);
}

// ---------------- small device helpers ----------------
__device__ __forceinline__ float eluf(float x) { return x > 0.f ? x : __expf(x) - 1.f; }

// order-preserving float -> uint monotone map (for atomic max of floats)
__device__ __forceinline__ unsigned fmono(float f) {
  unsigned u = __float_as_uint(f);
  return (u & 0x80000000u) ? ~u : (u | 0x80000000u);
}
__device__ __forceinline__ float funmono(unsigned u) {
  return (u & 0x80000000u) ? __uint_as_float(u & 0x7fffffffu) : __uint_as_float(~u);
}

// ---------------- utility fill kernels ----------------
__global__ void fill_f32_kernel(float* p, float v, int n) {
  int i = blockIdx.x * blockDim.x + threadIdx.x;
  if (i < n) p[i] = v;
}
__global__ void fill_i32_kernel(int* p, int v, int n) {
  int i = blockIdx.x * blockDim.x + threadIdx.x;
  if (i < n) p[i] = v;
}
__global__ void fill_negu_kernel(unsigned* p, int n) {
  int i = blockIdx.x * blockDim.x + threadIdx.x;
  if (i < n) p[i] = fmono(NEGF);
}

__global__ void init_nodes_kernel(const int* __restrict__ b, int* alive, int* bc) {
  int i = blockIdx.x * blockDim.x + threadIdx.x;
  if (i >= NODESN) return;
  alive[i] = 1;
  bc[i] = b[i];
}

// ---------------- input MLP: 5 -> 32 -> 64 -> 64 with elu ----------------
__global__ __launch_bounds__(256) void input_mlp_kernel(
    const float* __restrict__ x, const float* __restrict__ dn,
    const float* __restrict__ w1, const float* __restrict__ b1,
    const float* __restrict__ w2, const float* __restrict__ b2,
    const float* __restrict__ w3, const float* __restrict__ b3,
    float* __restrict__ h) {
  int i = blockIdx.x * blockDim.x + threadIdx.x;
  if (i >= NODESN) return;
  float v[5];
#pragma unroll
  for (int c = 0; c < 5; ++c) v[c] = x[i * 5 + c] * dn[c];
  float l1[32];
  for (int o = 0; o < 32; ++o) {
    float s = b1[o];
#pragma unroll
    for (int c = 0; c < 5; ++c) s += v[c] * w1[c * 32 + o];
    l1[o] = eluf(s);
  }
  float l2[64];
  for (int o = 0; o < 64; ++o) {
    float s = b2[o];
#pragma unroll
    for (int c = 0; c < 32; ++c) s += l1[c] * w2[c * 64 + o];
    l2[o] = eluf(s);
  }
  for (int o = 0; o < 64; ++o) {
    float s = b3[o];
#pragma unroll
    for (int c = 0; c < 64; ++c) s += l2[c] * w3[c * 64 + o];
    h[(size_t)i * HDIM + o] = eluf(s);
  }
}

// ---------------- per-level prep: bf16 cast + liveness-folded sq norms -----
// sq[i] = alive ? |x|^2 : BIG.  A dead endpoint then pushes the distance past
// the 0.5*BIG validity threshold, so the kNN inner loop needs no alive loads.
__global__ void prep_kernel(const float* __restrict__ h, const int* __restrict__ alive,
                            __bf16* __restrict__ hbf, float* __restrict__ sq) {
  int i = blockIdx.x * blockDim.x + threadIdx.x;
  if (i >= NODESN) return;
  float s = 0.f;
#pragma unroll
  for (int c0 = 0; c0 < HDIM; c0 += 4) {
    v4f_t v = *(const v4f_t*)(h + (size_t)i * HDIM + c0);
    s += v[0] * v[0] + v[1] * v[1] + v[2] * v[2] + v[3] * v[3];
    v4bf_t pk;
    pk[0] = (__bf16)v[0]; pk[1] = (__bf16)v[1];
    pk[2] = (__bf16)v[2]; pk[3] = (__bf16)v[3];
    *(v4bf_t*)(hbf + (size_t)i * HDIM + c0) = pk;
  }
  sq[i] = alive[i] ? s : BIGF;
}

// ---------------- kNN: WMMA distance tiles + register top-16 ----------------
// grid = NODESN/16 query tiles; block = 128 (4 waves). Each wave computes
// 16x16 dot-product tiles via 2x v_wmma_f32_16x16x32_bf16 (K=64).
__global__ __launch_bounds__(128) void knn_kernel(
    const __bf16* __restrict__ hbf, const float* __restrict__ sq,
    const int* __restrict__ alive, int* __restrict__ knn_idx,
    int* __restrict__ knn_ev) {
  __shared__ __bf16 qa[16 * HDIM];        // query tile, row-major MxK
  __shared__ __bf16 cbsw[4][2 * 512];     // per-wave candidate B-frags (swizzled)
  __shared__ float  dist[16 * CCH];

  int tid = threadIdx.x, lane = tid & 31, wave = tid >> 5;
  int q0 = blockIdx.x * 16;               // global query base
  int g = q0 / NPGC;
  int gbase = g * NPGC;

  // stage query tile (one v8bf per thread)
  {
    v8bf_t v = *(const v8bf_t*)(hbf + (size_t)q0 * HDIM + tid * 8);
    *(v8bf_t*)(qa + tid * 8) = v;
  }
  __syncthreads();

  v16bf a0 = load_a_frag(qa, HDIM, 0, lane);
  v16bf a1 = load_a_frag(qa, HDIM, 32, lane);

  float bd[KNB];
  int   bj[KNB];
#pragma unroll
  for (int k = 0; k < KNB; ++k) { bd[k] = BIGF; bj[k] = gbase; }

  int m_add = (lane & 16) ? 8 : 0;
  int ncol  = lane & 15;

  // per-lane query norms for the 8 accumulator rows (loop-invariant)
  float sqq[8];
#pragma unroll
  for (int r = 0; r < 8; ++r) sqq[r] = sq[q0 + r + m_add];

  for (int half = 0; half < NPGC / CCH; ++half) {
    for (int ct = wave; ct < CCH / 16; ct += 4) {
      int c0 = gbase + half * CCH + ct * 16;
      __builtin_prefetch(hbf + (size_t)(c0 + 16) * HDIM, 0, 1);
      // load candidate tile straight into swizzled B-fragment layout:
      // one v8bf global load -> one v8bf LDS store per 8 features.
#pragma unroll
      for (int gi = 0; gi < 4; ++gi) {
        int grp = lane + gi * 32;                 // 128 groups of 8 features
        int n = grp >> 3, f0 = (grp & 7) * 8;
        v8bf_t v = *(const v8bf_t*)(hbf + (size_t)(c0 + n) * HDIM + f0);
        int lsel = n + (((f0 & 31) >= 16) ? 16 : 0);
        *(v8bf_t*)(cbsw[wave] + (f0 >> 5) * 512 + lsel * 16 + (f0 & 15)) = v;
      }
      v16bf b0 = load_b_frag_sw(cbsw[wave], lane);
      v16bf b1 = load_b_frag_sw(cbsw[wave] + 512, lane);
      v8f acc = {};
      acc = wmma_bf16(a0, b0, acc);
      acc = wmma_bf16(a1, b1, acc);
      int cj = c0 + ncol;
      float sqc = sq[cj];                        // one load per tile per lane
#pragma unroll
      for (int r = 0; r < 8; ++r) {
        int qi = q0 + r + m_add;
        float d = sqq[r] + sqc - 2.0f * acc[r];
        dist[(r + m_add) * CCH + ct * 16 + ncol] = (qi == cj) ? BIGF : d;
      }
    }
    __syncthreads();
    if (tid < 16) {  // top-16 insertion per query row
      for (int j = 0; j < CCH; ++j) {
        float d = dist[tid * CCH + j];
        if (d < bd[KNB - 1]) {
          int gj = gbase + half * CCH + j;
          int p = KNB - 1;
          while (p > 0 && bd[p - 1] > d) {
            bd[p] = bd[p - 1]; bj[p] = bj[p - 1]; --p;
          }
          bd[p] = d; bj[p] = gj;
        }
      }
    }
    __syncthreads();
  }
  if (tid < 16) {
    int q = q0 + tid;
    int av = alive[q];
#pragma unroll
    for (int k = 0; k < KNB; ++k) {
      knn_idx[q * KNB + k] = bj[k];
      knn_ev[q * KNB + k] = (bd[k] < 0.5f * BIGF) && av;
    }
  }
}

// ---------------- undirected edges with mutual-dedup (to_undirected) -------
__global__ void build_edges_kernel(const int* __restrict__ knn_idx,
                                   const int* __restrict__ knn_ev,
                                   int* esrc, int* edst, int* evalid) {
  int e = blockIdx.x * blockDim.x + threadIdx.x;
  if (e >= EDIR) return;
  int i = e / KNB;
  int src = knn_idx[e];
  int v = knn_ev[e];
  esrc[e] = src; edst[e] = i; evalid[e] = v;
  // reverse edge (src <- i); drop if mutual (already present as forward of src)
  int keep = v;
  if (v) {
    for (int k = 0; k < KNB; ++k) {
      if (knn_idx[src * KNB + k] == i && knn_ev[src * KNB + k]) { keep = 0; break; }
    }
  }
  esrc[EDIR + e] = i; edst[EDIR + e] = src; evalid[EDIR + e] = keep;
}

// ---------------- EdgeConv: [16e x 128] @ W1[128x96] -> elu -> @ W2[96x64] --
// block = 128 (4 waves), 8 edge-tiles per wave; weights staged once per block
// in swizzled B-fragment layout (bf16). mid overlays feat (A-frags are in
// registers before mid is written; DS ops are in-order per wave).
__global__ __launch_bounds__(128) void edgeconv_kernel(
    const float* __restrict__ h, const int* __restrict__ esrc,
    const int* __restrict__ edst, const int* __restrict__ evalid,
    const float* __restrict__ w1, const float* __restrict__ b1,
    const float* __restrict__ w2, const float* __restrict__ b2,
    float* __restrict__ out) {
  __shared__ __bf16 w1sw[24 * 512];    // frag (nt,kk) = nt*4+kk, nt<6, kk<4
  __shared__ __bf16 w2sw[12 * 512];    // frag (nt,kk) = nt*3+kk, nt<4, kk<3
  __shared__ __bf16 fm[4][16 * 128];   // per-wave feat (mid overlays [0..1535])
  __shared__ int    dstn[4][16];
  __shared__ float  evm[4][16];

  int tid = threadIdx.x, lane = tid & 31, wave = tid >> 5;

  // stage W1 fragments: element (k = kk*32 + hi16 + v, n = nt*16 + (l&15))
  for (int t = tid; t < 24 * 512; t += 128) {
    int f = t >> 9, r = t & 511;
    int nt = f >> 2, kk = f & 3;
    int l = r >> 4, v = r & 15;
    int k = kk * 32 + ((l & 16) ? 16 : 0) + v;
    int n = nt * 16 + (l & 15);
    w1sw[t] = (__bf16)w1[k * 96 + n];
  }
  for (int t = tid; t < 12 * 512; t += 128) {
    int f = t >> 9, r = t & 511;
    int nt = f / 3, kk = f - nt * 3;
    int l = r >> 4, v = r & 15;
    int k = kk * 32 + ((l & 16) ? 16 : 0) + v;
    int n = nt * 16 + (l & 15);
    w2sw[t] = (__bf16)w2[k * 64 + n];
  }
  __syncthreads();

  __bf16* featp = fm[wave];
  int m_add = (lane & 16) ? 8 : 0;
  int ncol  = lane & 15;

  for (int it = 0; it < 8; ++it) {
    int tile = (blockIdx.x * 4 + wave) * 8 + it;
    int e0 = tile * 16;
    if (lane < 16) {
      dstn[wave][lane] = edst[e0 + lane];
      evm[wave][lane] = evalid[e0 + lane] ? 1.0f : 0.0f;
    }
    // gather [x_i, x_j - x_i] as bf16: float4 loads, packed v4bf stores
    for (int t = lane; t < 512; t += 32) {
      int e = t >> 5, c0 = (t & 31) * 4;
      int s = esrc[e0 + e], d = edst[e0 + e];
      v4f_t xd = *(const v4f_t*)(h + (size_t)d * HDIM + (c0 & 63));
      v4f_t val = xd;
      if (c0 >= 64) {
        v4f_t xs = *(const v4f_t*)(h + (size_t)s * HDIM + (c0 - 64));
        val = xs - xd;
      }
      v4bf_t pk;
      pk[0] = (__bf16)val[0]; pk[1] = (__bf16)val[1];
      pk[2] = (__bf16)val[2]; pk[3] = (__bf16)val[3];
      *(v4bf_t*)(featp + e * 128 + c0) = pk;
    }

    // layer 1: 16x128 @ 128x96
    v16bf a[4];
#pragma unroll
    for (int kk = 0; kk < 4; ++kk) a[kk] = load_a_frag(featp, 128, kk * 32, lane);
#pragma unroll
    for (int nt = 0; nt < 6; ++nt) {
      v8f acc = {};
#pragma unroll
      for (int kk = 0; kk < 4; ++kk) {
        v16bf b = load_b_frag_sw(w1sw + (nt * 4 + kk) * 512, lane);
        acc = wmma_bf16(a[kk], b, acc);
      }
#pragma unroll
      for (int r = 0; r < 8; ++r) {
        int m = r + m_add, ch = nt * 16 + ncol;
        featp[m * 96 + ch] = (__bf16)eluf(acc[r] + b1[ch]);  // mid overlay
      }
    }

    // layer 2: 16x96 @ 96x64
    v16bf a2[3];
#pragma unroll
    for (int kk = 0; kk < 3; ++kk) a2[kk] = load_a_frag(featp, 96, kk * 32, lane);
#pragma unroll
    for (int nt = 0; nt < 4; ++nt) {
      v8f acc = {};
#pragma unroll
      for (int kk = 0; kk < 3; ++kk) {
        v16bf b = load_b_frag_sw(w2sw + (nt * 3 + kk) * 512, lane);
        acc = wmma_bf16(a2[kk], b, acc);
      }
#pragma unroll
      for (int r = 0; r < 8; ++r) {
        int m = r + m_add, ch = nt * 16 + ncol;
        if (evm[wave][m] != 0.0f) {
          float v = eluf(acc[r] + b2[ch]);
          atomicAdd(&out[(size_t)dstn[wave][m] * HDIM + ch], v);
        }
      }
    }
  }
}

// ---------------- normalized cut ----------------
__global__ void edge_w_deg_kernel(const float* __restrict__ h2,
                                  const int* __restrict__ esrc,
                                  const int* __restrict__ edst,
                                  const int* __restrict__ ev,
                                  float* __restrict__ ew, float* __restrict__ deg) {
  int e = blockIdx.x * blockDim.x + threadIdx.x;
  if (e >= ETOT) return;
  int s = esrc[e], d = edst[e];
  float acc = 0.f;
#pragma unroll
  for (int c0 = 0; c0 < HDIM; c0 += 4) {
    v4f_t a = *(const v4f_t*)(h2 + (size_t)s * HDIM + c0);
    v4f_t b = *(const v4f_t*)(h2 + (size_t)d * HDIM + c0);
    v4f_t df = a - b;
    acc += df[0] * df[0] + df[1] * df[1] + df[2] * df[2] + df[3] * df[3];
  }
  ew[e] = sqrtf(acc);
  if (ev[e]) atomicAdd(&deg[d], 1.0f);
}

__global__ void inv_deg_kernel(const float* deg, float* inv, int n) {
  int i = blockIdx.x * blockDim.x + threadIdx.x;
  if (i < n) inv[i] = deg[i] > 0.f ? 1.0f / deg[i] : 0.0f;
}

__global__ void edge_cut_kernel(const float* __restrict__ ew,
                                const int* __restrict__ esrc,
                                const int* __restrict__ edst,
                                const int* __restrict__ ev,
                                const float* __restrict__ inv,
                                float* __restrict__ ecut) {
  int e = blockIdx.x * blockDim.x + threadIdx.x;
  if (e >= ETOT) return;
  ecut[e] = ev[e] ? ew[e] * (inv[esrc[e]] + inv[edst[e]]) : NEGF;
}

// ---------------- graclus heavy-edge matching (16 handshake iters) ---------
__global__ void grac_reset_kernel(unsigned* bwu, int* bnb, int n) {
  int i = blockIdx.x * blockDim.x + threadIdx.x;
  if (i < n) { bwu[i] = fmono(NEGF); bnb[i] = -1; }
}
__global__ void grac_bestw_kernel(const float* __restrict__ ecut,
                                  const int* __restrict__ esrc,
                                  const int* __restrict__ edst,
                                  const int* __restrict__ ev,
                                  const int* __restrict__ match,
                                  unsigned* __restrict__ bwu) {
  int e = blockIdx.x * blockDim.x + threadIdx.x;
  if (e >= ETOT) return;
  int s = esrc[e], d = edst[e];
  bool un = ev[e] && (match[s] < 0) && (match[d] < 0);
  float we = un ? ecut[e] : NEGF;
  atomicMax(&bwu[d], fmono(we));
}
__global__ void grac_bestnb_kernel(const float* __restrict__ ecut,
                                   const int* __restrict__ esrc,
                                   const int* __restrict__ edst,
                                   const int* __restrict__ ev,
                                   const int* __restrict__ match,
                                   const unsigned* __restrict__ bwu,
                                   int* __restrict__ bnb) {
  int e = blockIdx.x * blockDim.x + threadIdx.x;
  if (e >= ETOT) return;
  int s = esrc[e], d = edst[e];
  bool un = ev[e] && (match[s] < 0) && (match[d] < 0);
  float we = un ? ecut[e] : NEGF;
  if (we > 0.5f * NEGF && fmono(we) == bwu[d]) atomicMax(&bnb[d], s);
}
__global__ void grac_prop_kernel(const int* match, const unsigned* bwu,
                                 const int* bnb, int* prop, int n) {
  int i = blockIdx.x * blockDim.x + threadIdx.x;
  if (i < n) prop[i] = (match[i] < 0 && funmono(bwu[i]) > 0.5f * NEGF) ? bnb[i] : -1;
}
__global__ void grac_match_kernel(const int* __restrict__ prop, int* match, int n) {
  int i = blockIdx.x * blockDim.x + threadIdx.x;
  if (i >= n) return;
  int p = prop[i];
  int pc = p < 0 ? 0 : (p > n - 1 ? n - 1 : p);
  if (p >= 0 && prop[pc] == i) match[i] = p;
}
__global__ void cluster_kernel(const int* match, int* clus, int n) {
  int i = blockIdx.x * blockDim.x + threadIdx.x;
  if (i >= n) return;
  int m = match[i];
  clus[i] = (m >= 0) ? (m < i ? m : i) : i;
}

// ---------------- cluster max-pool (atomic monotone-uint max) --------------
__global__ void pool_scatter_kernel(const float* __restrict__ h2,
                                    const int* __restrict__ alive,
                                    const int* __restrict__ clus,
                                    const int* __restrict__ bc,
                                    unsigned* __restrict__ pool_u,
                                    int* __restrict__ alive2,
                                    int* __restrict__ batch2) {
  int t = blockIdx.x * blockDim.x + threadIdx.x;
  if (t >= NODESN * HDIM) return;
  int i = t >> 6, c = t & 63;
  if (!alive[i]) return;
  int cl = clus[i];
  atomicMax(&pool_u[(size_t)cl * HDIM + c], fmono(h2[t]));
  if (c == 0) {
    atomicOr(&alive2[cl], 1);
    atomicMax(&batch2[cl], bc[i]);
  }
}
__global__ void pool_finalize_kernel(const unsigned* __restrict__ pool_u,
                                     const int* __restrict__ alive2,
                                     const int* __restrict__ batch2,
                                     float* __restrict__ h, int* __restrict__ alive,
                                     int* __restrict__ bc) {
  int t = blockIdx.x * blockDim.x + threadIdx.x;
  if (t >= NODESN * HDIM) return;
  int i = t >> 6, c = t & 63;
  int a2 = alive2[i];
  h[t] = a2 ? funmono(pool_u[t]) : 0.0f;
  if (c == 0) { alive[i] = a2; bc[i] = batch2[i]; }
}

// ---------------- global per-graph max pool + output head ------------------
__global__ void gpool_scatter_kernel(const float* __restrict__ h,
                                     const int* __restrict__ alive,
                                     const int* __restrict__ bc,
                                     unsigned* __restrict__ gpool_u) {
  int t = blockIdx.x * blockDim.x + threadIdx.x;
  if (t >= NODESN * HDIM) return;
  int i = t >> 6, c = t & 63;
  if (!alive[i]) return;
  int g = bc[i];
  if (g < 0 || g >= BG) return;
  atomicMax(&gpool_u[(size_t)g * HDIM + c], fmono(h[t]));
}

__global__ __launch_bounds__(32) void out_mlp_kernel(
    const unsigned* __restrict__ gpool_u,
    const float* __restrict__ w1, const float* __restrict__ b1,
    const float* __restrict__ w2, const float* __restrict__ b2,
    const float* __restrict__ w3, const float* __restrict__ b3,
    float* __restrict__ out) {
  int t = threadIdx.x;
  if (t >= BG) return;
  float g[64], l1[64], l2[32];
  for (int c = 0; c < 64; ++c) g[c] = funmono(gpool_u[t * 64 + c]);
  for (int o = 0; o < 64; ++o) {
    float s = b1[o];
#pragma unroll
    for (int c = 0; c < 64; ++c) s += g[c] * w1[c * 64 + o];
    l1[o] = eluf(s);
  }
  for (int o = 0; o < 32; ++o) {
    float s = b2[o];
#pragma unroll
    for (int c = 0; c < 64; ++c) s += l1[c] * w2[c * 32 + o];
    l2[o] = eluf(s);
  }
  float s = b3[0];
#pragma unroll
  for (int c = 0; c < 32; ++c) s += l2[c] * w3[c];
  out[t] = s;
}

// ---------------- host launch ----------------
extern "C" void kernel_launch(void* const* d_in, const int* in_sizes, int n_in,
                              void* d_out, int out_size, void* d_ws, size_t ws_size,
                              hipStream_t stream) {
  (void)in_sizes; (void)n_in; (void)out_size; (void)ws_size;
  const float* x        = (const float*)d_in[0];
  const int*   batch_in = (const int*)d_in[1];
  const float* datanorm = (const float*)d_in[2];
  const float* in_w1 = (const float*)d_in[3];  const float* in_b1 = (const float*)d_in[4];
  const float* in_w2 = (const float*)d_in[5];  const float* in_b2 = (const float*)d_in[6];
  const float* in_w3 = (const float*)d_in[7];  const float* in_b3 = (const float*)d_in[8];
  const float* c1_w1 = (const float*)d_in[9];  const float* c1_b1 = (const float*)d_in[10];
  const float* c1_w2 = (const float*)d_in[11]; const float* c1_b2 = (const float*)d_in[12];
  const float* c2_w1 = (const float*)d_in[13]; const float* c2_b1 = (const float*)d_in[14];
  const float* c2_w2 = (const float*)d_in[15]; const float* c2_b2 = (const float*)d_in[16];
  const float* o_w1  = (const float*)d_in[17]; const float* o_b1  = (const float*)d_in[18];
  const float* o_w2  = (const float*)d_in[19]; const float* o_b2  = (const float*)d_in[20];
  const float* o_w3  = (const float*)d_in[21]; const float* o_b3  = (const float*)d_in[22];

  unsigned char* ws = (unsigned char*)d_ws;
  size_t off = 0;
  auto take = [&](size_t bytes) -> void* {
    void* p = ws + off;
    off += (bytes + 255) & ~(size_t)255;
    return p;
  };
  float*    h       = (float*)take((size_t)NODESN * HDIM * 4);
  __bf16*   hbf     = (__bf16*)take((size_t)NODESN * HDIM * 2);
  float*    sq      = (float*)take((size_t)NODESN * 4);
  int*      alive   = (int*)take((size_t)NODESN * 4);
  int*      bc      = (int*)take((size_t)NODESN * 4);
  float*    h2      = (float*)take((size_t)NODESN * HDIM * 4);
  int*      knn_idx = (int*)take((size_t)NODESN * KNB * 4);
  int*      knn_ev  = (int*)take((size_t)NODESN * KNB * 4);
  int*      esrc    = (int*)take((size_t)ETOT * 4);
  int*      edst    = (int*)take((size_t)ETOT * 4);
  int*      eval_   = (int*)take((size_t)ETOT * 4);
  float*    ew      = (float*)take((size_t)ETOT * 4);
  float*    ecut    = (float*)take((size_t)ETOT * 4);
  float*    deg     = (float*)take((size_t)NODESN * 4);
  float*    invd    = (float*)take((size_t)NODESN * 4);
  unsigned* bwu     = (unsigned*)take((size_t)NODESN * 4);
  int*      bnb     = (int*)take((size_t)NODESN * 4);
  int*      prop    = (int*)take((size_t)NODESN * 4);
  int*      match   = (int*)take((size_t)NODESN * 4);
  int*      clus    = (int*)take((size_t)NODESN * 4);
  unsigned* pool_u  = (unsigned*)take((size_t)NODESN * HDIM * 4);
  int*      alive2  = (int*)take((size_t)NODESN * 4);
  int*      batch2  = (int*)take((size_t)NODESN * 4);
  unsigned* gpool_u = (unsigned*)take((size_t)BG * HDIM * 4);

  const int NT = NODESN * HDIM;
  init_nodes_kernel<<<CDIV(NODESN, 256), 256, 0, stream>>>(batch_in, alive, bc);
  input_mlp_kernel<<<CDIV(NODESN, 256), 256, 0, stream>>>(
      x, datanorm, in_w1, in_b1, in_w2, in_b2, in_w3, in_b3, h);

  auto run_level = [&](const float* cw1, const float* cb1,
                       const float* cw2, const float* cb2) {
    prep_kernel<<<CDIV(NODESN, 256), 256, 0, stream>>>(h, alive, hbf, sq);
    knn_kernel<<<NODESN / 16, 128, 0, stream>>>(hbf, sq, alive, knn_idx, knn_ev);
    build_edges_kernel<<<CDIV(EDIR, 256), 256, 0, stream>>>(knn_idx, knn_ev, esrc, edst, eval_);
    fill_f32_kernel<<<CDIV(NT, 256), 256, 0, stream>>>(h2, 0.f, NT);
    edgeconv_kernel<<<ETOT / (16 * 32), 128, 0, stream>>>(h, esrc, edst, eval_,
                                                          cw1, cb1, cw2, cb2, h2);
    fill_f32_kernel<<<CDIV(NODESN, 256), 256, 0, stream>>>(deg, 0.f, NODESN);
    edge_w_deg_kernel<<<CDIV(ETOT, 256), 256, 0, stream>>>(h2, esrc, edst, eval_, ew, deg);
    inv_deg_kernel<<<CDIV(NODESN, 256), 256, 0, stream>>>(deg, invd, NODESN);
    edge_cut_kernel<<<CDIV(ETOT, 256), 256, 0, stream>>>(ew, esrc, edst, eval_, invd, ecut);
    fill_i32_kernel<<<CDIV(NODESN, 256), 256, 0, stream>>>(match, -1, NODESN);
    for (int it = 0; it < 16; ++it) {
      grac_reset_kernel<<<CDIV(NODESN, 256), 256, 0, stream>>>(bwu, bnb, NODESN);
      grac_bestw_kernel<<<CDIV(ETOT, 256), 256, 0, stream>>>(ecut, esrc, edst, eval_, match, bwu);
      grac_bestnb_kernel<<<CDIV(ETOT, 256), 256, 0, stream>>>(ecut, esrc, edst, eval_, match, bwu, bnb);
      grac_prop_kernel<<<CDIV(NODESN, 256), 256, 0, stream>>>(match, bwu, bnb, prop, NODESN);
      grac_match_kernel<<<CDIV(NODESN, 256), 256, 0, stream>>>(prop, match, NODESN);
    }
    cluster_kernel<<<CDIV(NODESN, 256), 256, 0, stream>>>(match, clus, NODESN);
    fill_negu_kernel<<<CDIV(NT, 256), 256, 0, stream>>>(pool_u, NT);
    fill_i32_kernel<<<CDIV(NODESN, 256), 256, 0, stream>>>(alive2, 0, NODESN);
    fill_i32_kernel<<<CDIV(NODESN, 256), 256, 0, stream>>>(batch2, -1, NODESN);
    pool_scatter_kernel<<<CDIV(NT, 256), 256, 0, stream>>>(h2, alive, clus, bc, pool_u, alive2, batch2);
    pool_finalize_kernel<<<CDIV(NT, 256), 256, 0, stream>>>(pool_u, alive2, batch2, h, alive, bc);
  };

  run_level(c1_w1, c1_b1, c1_w2, c1_b2);
  run_level(c2_w1, c2_b1, c2_w2, c2_b2);

  fill_negu_kernel<<<CDIV(BG * HDIM, 256), 256, 0, stream>>>(gpool_u, BG * HDIM);
  gpool_scatter_kernel<<<CDIV(NT, 256), 256, 0, stream>>>(h, alive, bc, gpool_u);
  out_mlp_kernel<<<1, 32, 0, stream>>>(gpool_u, o_w1, o_b1, o_w2, o_b2, o_w3, o_b3,
                                       (float*)d_out);
}
